// GraphEncoder_3556232921556
// MI455X (gfx1250) — matrix operand order repaired and checked by reading, not compiled, not run on hardware
//
#include <hip/hip_runtime.h>
#include <hip/hip_bf16.h>
#include <stdint.h>
#include <stddef.h>

typedef __attribute__((ext_vector_type(16))) __bf16 v16bf;
typedef __attribute__((ext_vector_type(8)))  __bf16 v8bf;
typedef __attribute__((ext_vector_type(8)))  float  v8f;

#define NN     50000
#define NE     800000
#define NG     256
#define TPB    256

// ---- helpers ---------------------------------------------------------------

__device__ __forceinline__ __bf16 f2bf(float f) {
  unsigned u = __builtin_bit_cast(unsigned, f);
  unsigned r = u + 0x7FFFu + ((u >> 16) & 1u);     // round-to-nearest-even
  unsigned short h = (unsigned short)(r >> 16);
  return __builtin_bit_cast(__bf16, h);
}

// ---- small elementwise kernels --------------------------------------------

__global__ void convert_bf(const float* __restrict__ x, __bf16* __restrict__ o, int n) {
  int i = blockIdx.x * TPB + threadIdx.x;
  if (i < n) o[i] = f2bf(x[i]);
}

__global__ void zero_f(float* p, int n) {
  int i = blockIdx.x * TPB + threadIdx.x;
  if (i < n) p[i] = 0.0f;
}

// Pack W[K,N] (f32 row-major) into per-lane WMMA B-operand order:
// Wp[((kt*ntiles + nt)*32 + lane)*16 + j] = W[kt*32 + (j&7) + (j>=8?16:0) + (lane>=16?8:0)][nt*16 + lane%16]
__global__ void pack_w(const float* __restrict__ W, __bf16* __restrict__ Wp, int K, int N) {
  int p = blockIdx.x * TPB + threadIdx.x;
  if (p >= K * N) return;
  int j      = p & 15;
  int lane   = (p >> 4) & 31;
  int ntiles = N >> 4;
  int nt     = (p >> 9) % ntiles;
  int kt     = p / (512 * ntiles);
  int k = kt * 32 + (j & 7) + ((j >> 3) << 4) + ((lane >> 4) ? 8 : 0);
  int n = nt * 16 + (lane & 15);
  Wp[p] = f2bf(W[(size_t)k * N + n]);
}

// ---- degree / norm ---------------------------------------------------------

__global__ void deg_init(float* deg) {
  int i = blockIdx.x * TPB + threadIdx.x;
  if (i < NN) deg[i] = 1.0f;                 // self-loop
}
__global__ void deg_edges(const long long* __restrict__ dst, float* deg) {
  int e = blockIdx.x * TPB + threadIdx.x;
  if (e < NE) atomicAdd(&deg[(int)dst[e]], 1.0f);
}
__global__ void dinv_k(const float* __restrict__ deg, float* __restrict__ dinv) {
  int i = blockIdx.x * TPB + threadIdx.x;
  if (i < NN) dinv[i] = (deg[i] > 0.0f) ? rsqrtf(deg[i]) : 0.0f;
}

// ---- WMMA GEMM -------------------------------------------------------------
// Y[M,N] = A[M,K](bf16) @ Wp(packed bf16); optional bias/relu epilogue,
// optional fused self-loop write Hself = Y * dinv[row]^2, optional bf16 copy.
// block = 256 threads (8 waves); each wave: one 16x16 tile; grid = (M/16, N/128).
// A-tile (16 x K) staged once per block in LDS via async global->LDS loads.

__global__ void wmma_gemm(const __bf16* __restrict__ A, const __bf16* __restrict__ Wp,
                          const float* __restrict__ bias, const float* __restrict__ dinv,
                          float* __restrict__ Yf, float* __restrict__ Hself,
                          __bf16* __restrict__ Ybf, int M, int K, int N, int relu) {
  __shared__ __align__(16) __bf16 sA[16 * (256 + 8)];   // max K=256, +8 elem row pad

  int tid     = threadIdx.x;
  int lane    = tid & 31;
  int wave    = tid >> 5;
  int rowBase = blockIdx.x * 16;
  int ntiles  = N >> 4;
  int ntile   = blockIdx.y * 8 + wave;
  int colBase = ntile * 16;
  int laneHi  = lane >> 4;
  int stride  = K + 8;                      // padded LDS row stride (elements)

  // --- async stage A tile: 16 rows x K bf16, 16B chunks, one per lane ------
  int chunksPerRow = K >> 3;                // 8 bf16 per 16B chunk
  int totalChunks  = chunksPerRow << 4;     // 256 (K=128) or 512 (K=256)
  unsigned ldsBase = (unsigned)(size_t)(&sA[0]);
  for (int c = tid; c < totalChunks; c += TPB) {
    int r    = c / chunksPerRow;
    int col8 = c - r * chunksPerRow;
    const __bf16* g = A + (size_t)(rowBase + r) * K + col8 * 8;
    unsigned l = ldsBase + (unsigned)((r * stride + col8 * 8) * 2);
    asm volatile("global_load_async_to_lds_b128 %0, %1, off"
                 :: "v"(l), "v"(g) : "memory");
  }
  asm volatile("s_wait_asynccnt 0x0" ::: "memory");
  __syncthreads();

  // --- WMMA main loop ------------------------------------------------------
  v8f c = {};
  int ktc = K >> 5;
  const __bf16* rowp = &sA[(lane & 15) * stride];
  for (int kt = 0; kt < ktc; ++kt) {
    int k0 = kt * 32 + (laneHi ? 8 : 0);
    v8bf alo = *reinterpret_cast<const v8bf*>(rowp + k0);        // ds_load_b128
    v8bf ahi = *reinterpret_cast<const v8bf*>(rowp + k0 + 16);   // ds_load_b128
    v16bf a;
#pragma unroll
    for (int i = 0; i < 8; ++i) { a[i] = alo[i]; a[i + 8] = ahi[i]; }
    v16bf b = *reinterpret_cast<const v16bf*>(                    // 32B/lane, packed
        Wp + ((size_t)(kt * ntiles + ntile) * 32 + lane) * 16);
    c = __builtin_amdgcn_wmma_f32_16x16x32_bf16(false, a, false, b,
                                                (short)0, c, false, false);
  }

  // --- epilogue ------------------------------------------------------------
  int col = colBase + (lane & 15);
  float bv = bias ? bias[col] : 0.0f;
#pragma unroll
  for (int r = 0; r < 8; ++r) {
    int row = rowBase + r + laneHi * 8;    // C/D layout: lanes16-31 => M+8
    float v = c[r] + bv;
    if (relu) v = fmaxf(v, 0.0f);
    size_t o = (size_t)row * N + col;
    if (Yf)    Yf[o] = v;
    if (Hself) { float di = dinv[row]; Hself[o] = v * di * di; }  // self-loop msg
    if (Ybf)   Ybf[o] = f2bf(v);
  }
}

// ---- aggregation -----------------------------------------------------------

__global__ void edge_scatter(const long long* __restrict__ src, const long long* __restrict__ dst,
                             const float* __restrict__ dinv, const float* __restrict__ Y,
                             float* __restrict__ H2) {
  int e = blockIdx.x * 8 + (threadIdx.x >> 5);   // 8 edges per 256-thread block
  if (e >= NE) return;
  int lane = threadIdx.x & 31;
  int s = (int)src[e], d = (int)dst[e];
  float nrm = dinv[s] * dinv[d];
  const float* ys = Y + (size_t)s * 128;
  __builtin_prefetch(ys, 0, 0);                  // global_prefetch_b8
  float4 v = reinterpret_cast<const float4*>(ys)[lane];
  float* hd = H2 + (size_t)d * 128 + lane * 4;
  atomicAdd(hd + 0, v.x * nrm);
  atomicAdd(hd + 1, v.y * nrm);
  atomicAdd(hd + 2, v.z * nrm);
  atomicAdd(hd + 3, v.w * nrm);
}

__global__ void bias_relu_bf(const float* __restrict__ H2, const float* __restrict__ b,
                             __bf16* __restrict__ Xbf) {
  int idx = blockIdx.x * TPB + threadIdx.x;
  if (idx >= NN * 128) return;
  float v = H2[idx] + b[idx & 127];
  Xbf[idx] = f2bf(fmaxf(v, 0.0f));
}

// ---- pooling ---------------------------------------------------------------

__global__ void pool_cnt(const long long* __restrict__ batch, float* cnts) {
  int i = blockIdx.x * TPB + threadIdx.x;
  if (i < NN) atomicAdd(&cnts[(int)batch[i]], 1.0f);
}
__global__ void pool_sum(const float* __restrict__ H2, const float* __restrict__ b3,
                         const long long* __restrict__ batch, float* sums) {
  int idx = blockIdx.x * TPB + threadIdx.x;
  if (idx >= NN * 128) return;
  int i = idx >> 7, ch = idx & 127;
  atomicAdd(&sums[(size_t)(int)batch[i] * 128 + ch], H2[idx] + b3[ch]);
}
__global__ void pool_fin(const float* __restrict__ sums, const float* __restrict__ cnts,
                         __bf16* __restrict__ pooledbf) {
  int idx = blockIdx.x * TPB + threadIdx.x;
  if (idx >= NG * 128) return;
  float cnt = fmaxf(cnts[idx >> 7], 1.0f);
  pooledbf[idx] = f2bf(sums[idx] / cnt);
}

// ---- driver ----------------------------------------------------------------

extern "C" void kernel_launch(void* const* d_in, const int* in_sizes, int n_in,
                              void* d_out, int out_size, void* d_ws, size_t ws_size,
                              hipStream_t stream) {
  (void)in_sizes; (void)n_in; (void)out_size; (void)ws_size;

  const float*     x    = (const float*)d_in[0];
  const long long* ei   = (const long long*)d_in[1];
  const long long* batch= (const long long*)d_in[2];
  const float* W1 = (const float*)d_in[3];  const float* b1 = (const float*)d_in[4];
  const float* W2 = (const float*)d_in[5];  const float* b2 = (const float*)d_in[6];
  const float* W3 = (const float*)d_in[7];  const float* b3 = (const float*)d_in[8];
  const float* Wm1= (const float*)d_in[9];  const float* bm1= (const float*)d_in[10];
  const float* Wm2= (const float*)d_in[11]; const float* bm2= (const float*)d_in[12];
  float* out = (float*)d_out;

  const long long* src = ei;
  const long long* dst = ei + NE;

  char* ws = (char*)d_ws;
  size_t off = 0;
  auto wsAlloc = [&](size_t bytes) -> void* {
    void* p = (void*)(ws + off);
    off += (bytes + 255) & ~(size_t)255;
    return p;
  };
  __bf16* Xbf   = (__bf16*)wsAlloc((size_t)NN * 128 * 2);
  float*  Y     = (float*) wsAlloc((size_t)NN * 128 * 4);
  float*  H2    = (float*) wsAlloc((size_t)NN * 128 * 4);
  float*  deg   = (float*) wsAlloc((size_t)NN * 4);
  float*  dinv  = (float*) wsAlloc((size_t)NN * 4);
  __bf16* W1p   = (__bf16*)wsAlloc(128 * 128 * 2);
  __bf16* W2p   = (__bf16*)wsAlloc(128 * 128 * 2);
  __bf16* W3p   = (__bf16*)wsAlloc(128 * 128 * 2);
  __bf16* Wm1p  = (__bf16*)wsAlloc(128 * 256 * 2);
  __bf16* Wm2p  = (__bf16*)wsAlloc(256 * 768 * 2);
  float*  sums  = (float*) wsAlloc(NG * 128 * 4);
  float*  cnts  = (float*) wsAlloc(NG * 4);
  __bf16* poolb = (__bf16*)wsAlloc(NG * 128 * 2);
  __bf16* mlp1b = (__bf16*)wsAlloc(NG * 256 * 2);

  const int nodeElems = NN * 128;                  // 6,400,000
  const int gbN  = (NN + TPB - 1) / TPB;           // 196
  const int gbE  = (NE + TPB - 1) / TPB;           // 3125
  const int gbNE = nodeElems / TPB;                // 25000 (exact)

  // input features -> bf16, weight packing
  convert_bf<<<gbNE, TPB, 0, stream>>>(x, Xbf, nodeElems);
  pack_w<<<(128 * 128 + TPB - 1) / TPB, TPB, 0, stream>>>(W1,  W1p,  128, 128);
  pack_w<<<(128 * 128 + TPB - 1) / TPB, TPB, 0, stream>>>(W2,  W2p,  128, 128);
  pack_w<<<(128 * 128 + TPB - 1) / TPB, TPB, 0, stream>>>(W3,  W3p,  128, 128);
  pack_w<<<(128 * 256 + TPB - 1) / TPB, TPB, 0, stream>>>(Wm1, Wm1p, 128, 256);
  pack_w<<<(256 * 768 + TPB - 1) / TPB, TPB, 0, stream>>>(Wm2, Wm2p, 256, 768);

  // symmetric normalization terms (deg includes self-loop)
  deg_init<<<gbN, TPB, 0, stream>>>(deg);
  deg_edges<<<gbE, TPB, 0, stream>>>(dst, deg);
  dinv_k<<<gbN, TPB, 0, stream>>>(deg, dinv);

  // 3 GCN layers (self-loop message fused into GEMM epilogue)
  const __bf16* Wpacks[3] = {W1p, W2p, W3p};
  const float*  biases[3] = {b1, b2, b3};
  for (int l = 0; l < 3; ++l) {
    wmma_gemm<<<dim3(NN / 16, 1), TPB, 0, stream>>>(Xbf, Wpacks[l], nullptr, dinv,
                                                    Y, H2, nullptr, NN, 128, 128, 0);
    edge_scatter<<<NE / 8, TPB, 0, stream>>>(src, dst, dinv, Y, H2);
    if (l < 2)
      bias_relu_bf<<<gbNE, TPB, 0, stream>>>(H2, biases[l], Xbf);
    // layer 3: bias fused into pooling, no relu (matches reference)
  }

  // global mean pool (bias b3 added per-element inside pool_sum)
  zero_f<<<(NG * 128 + TPB - 1) / TPB, TPB, 0, stream>>>(sums, NG * 128);
  zero_f<<<1, TPB, 0, stream>>>(cnts, NG);
  pool_cnt<<<gbN, TPB, 0, stream>>>(batch, cnts);
  pool_sum<<<gbNE, TPB, 0, stream>>>(H2, b3, batch, sums);
  pool_fin<<<(NG * 128 + TPB - 1) / TPB, TPB, 0, stream>>>(sums, cnts, poolb);

  // MLP head: [256,128]@[128,256]+relu -> [256,256]@[256,768]
  wmma_gemm<<<dim3(NG / 16, 256 / 128), TPB, 0, stream>>>(poolb, Wm1p, bm1, nullptr,
                                                          nullptr, nullptr, mlp1b,
                                                          NG, 128, 256, 1);
  wmma_gemm<<<dim3(NG / 16, 768 / 128), TPB, 0, stream>>>(mlp1b, Wm2p, bm2, nullptr,
                                                          out, nullptr, nullptr,
                                                          NG, 256, 768, 0);
}
